// UnifiedQuantumEnhancements_51857435131904
// MI455X (gfx1250) — matrix operand order repaired and checked by reading, not compiled, not run on hardware
//
#include <hip/hip_runtime.h>
#include <hip/hip_bf16.h>
#include <math.h>

typedef __attribute__((ext_vector_type(16))) __bf16 v16bf;
typedef __attribute__((ext_vector_type(8)))  float  v8f;
typedef __attribute__((ext_vector_type(4)))  unsigned int u32x4;
typedef __attribute__((ext_vector_type(8)))  int          i32x8;
typedef __attribute__((ext_vector_type(4)))  int          i32x4;

#define D_DIM 2048
#define L_DIM 2048
#define B_DIM 4
#define S_SPLIT 16
#define DT_CONST 0.01f

#if defined(__gfx1250__) && __has_builtin(__builtin_amdgcn_tensor_load_to_lds) && \
    __has_builtin(__builtin_amdgcn_s_wait_tensorcnt)
#define USE_TDM 1
#else
#define USE_TDM 0
#endif

// ---------------------------------------------------------------------------
// Kernel A1: partial column sums of x over a slice of L (b128 streaming).
// part[s][b][d] = sum_{l in slice s} x[b][l][d]
// ---------------------------------------------------------------------------
__global__ void colsum_partial(const float* __restrict__ x, float* __restrict__ part) {
    const int b  = blockIdx.x;            // 0..3
    const int dc = blockIdx.y;            // 0..1  (1024 d's each)
    const int s  = blockIdx.z;            // 0..15 (128 l's each)
    const int t  = threadIdx.x;           // 0..255
    const int q  = dc * 1024 + t * 4;     // d base; 4 consecutive floats/thread
    const int LROWS = L_DIM / S_SPLIT;    // 128
    const float4* base =
        (const float4*)(x + ((size_t)b * L_DIM + (size_t)s * LROWS) * D_DIM + q);
    float ax = 0.f, ay = 0.f, az = 0.f, aw = 0.f;
    #pragma unroll 4
    for (int l = 0; l < LROWS; ++l) {
        const float4 xv = base[(size_t)l * (D_DIM / 4)];
        ax += xv.x; ay += xv.y; az += xv.z; aw += xv.w;
    }
    float4 r; r.x = ax; r.y = ay; r.z = az; r.w = aw;
    *(float4*)(part + ((size_t)s * B_DIM + b) * D_DIM + q) = r;
}

// ---------------------------------------------------------------------------
// Kernel A2: finish reduction, scale by 1/L -> x_mean[b][d]
// ---------------------------------------------------------------------------
__global__ void colsum_finish(const float* __restrict__ part, float* __restrict__ x_mean) {
    const int q4 = blockIdx.x * 256 + threadIdx.x;   // float4 index, 0..2047
    float ax = 0.f, ay = 0.f, az = 0.f, aw = 0.f;
    #pragma unroll
    for (int s = 0; s < S_SPLIT; ++s) {
        const float4 p = ((const float4*)part)[(size_t)s * (B_DIM * D_DIM / 4) + q4];
        ax += p.x; ay += p.y; az += p.z; aw += p.w;
    }
    const float sc = 1.0f / (float)L_DIM;
    float4 r; r.x = ax * sc; r.y = ay * sc; r.z = az * sc; r.w = aw * sc;
    ((float4*)x_mean)[q4] = r;
}

// ---------------------------------------------------------------------------
// WMMA bf16 VGPR-layout helpers (ISA 7.12.2), wave32:
//  A (16x32): lane = half*16 + m; element e -> K = ((e<8)?e:e+8) + 8*half
//  B (32x16): lane = half*16 + n; element e -> K = ((e<8)?e:e+8) + 8*half
//  C/D (16x16 f32): lane = half*16 + n holds rows M = r + 8*half
// ---------------------------------------------------------------------------
__device__ __forceinline__ int a_k_of(int e, int half) {
    return ((e < 8) ? e : e + 8) + 8 * half;
}
__device__ __forceinline__ int b_k_of(int e, int half) {
    return ((e < 8) ? e : e + 8) + 8 * half;
}

// ---------------------------------------------------------------------------
// Kernel B: g[b][i] = sum_j h_proj[i][j] * x_mean[b][j]   (one wave / 16 rows)
// ---------------------------------------------------------------------------
__global__ void gemm1_wmma(const float* __restrict__ h_proj,
                           const float* __restrict__ x_mean,
                           float* __restrict__ g) {
    const int i0   = blockIdx.x * 16;
    const int lane = threadIdx.x;
    const int half = lane >> 4;
    const int mn   = lane & 15;
    const float bsel = (mn < B_DIM) ? 1.0f : 0.0f;
    const int   brow = (mn & 3);

    v8f acc = {};
    for (int k0 = 0; k0 < D_DIM; k0 += 32) {
        v16bf a, b;
        #pragma unroll
        for (int e = 0; e < 16; ++e) {
            const int ka = a_k_of(e, half);
            a[e] = (__bf16)h_proj[(size_t)(i0 + mn) * D_DIM + (k0 + ka)];
            const int kb = b_k_of(e, half);
            b[e] = (__bf16)(x_mean[(size_t)brow * D_DIM + (k0 + kb)] * bsel);
        }
        acc = __builtin_amdgcn_wmma_f32_16x16x32_bf16(
                  false, a, false, b, (short)0, acc, false, false);
    }
    if (mn < B_DIM) {
        #pragma unroll
        for (int r = 0; r < 8; ++r) {
            const int m = r + 8 * half;
            g[(size_t)mn * D_DIM + (i0 + m)] = acc[r];
        }
    }
}

// ---------------------------------------------------------------------------
// Kernel C: delta[b][d] = DT*( sum_k g[b][k]*(j_upper[d][k]-j_upper[k][d])
//                              - g[b][d]*softplus(r_diag[d]) )
// ---------------------------------------------------------------------------
__global__ void gemm2_wmma(const float* __restrict__ j_upper,
                           const float* __restrict__ r_diag,
                           const float* __restrict__ g,
                           float* __restrict__ delta) {
    const int d0   = blockIdx.x * 16;
    const int lane = threadIdx.x;
    const int half = lane >> 4;
    const int mn   = lane & 15;
    const float msel = (mn < B_DIM) ? 1.0f : 0.0f;
    const int   grow = (mn & 3);

    v8f acc = {};
    for (int k0 = 0; k0 < D_DIM; k0 += 32) {
        v16bf a, b;
        #pragma unroll
        for (int e = 0; e < 16; ++e) {
            const int ka = a_k_of(e, half);
            a[e] = (__bf16)(g[(size_t)grow * D_DIM + (k0 + ka)] * msel);
            const int kb = b_k_of(e, half);
            const int k  = k0 + kb;
            const float jt = j_upper[(size_t)(d0 + mn) * D_DIM + k];
            const float jn = j_upper[(size_t)k * D_DIM + (d0 + mn)];
            b[e] = (__bf16)(jt - jn);
        }
        acc = __builtin_amdgcn_wmma_f32_16x16x32_bf16(
                  false, a, false, b, (short)0, acc, false, false);
    }
    const int dcol = d0 + mn;
    const float sp = log1pf(expf(r_diag[dcol]));
    if (half == 0) {
        #pragma unroll
        for (int r = 0; r < B_DIM; ++r) {
            const float diag = g[(size_t)r * D_DIM + dcol] * sp;
            delta[(size_t)r * D_DIM + dcol] = DT_CONST * (acc[r] - diag);
        }
    }
}

// ---------------------------------------------------------------------------
// Degree-8 Chebyshev evaluation
// ---------------------------------------------------------------------------
__device__ __forceinline__ float cheb8(float xn, const float* c) {
    float tp = 1.0f, tc = xn;
    float o = c[0] + c[1] * xn;
    #pragma unroll
    for (int i = 2; i <= 8; ++i) {
        const float tn = 2.0f * xn * tc - tp;
        o += c[i] * tn;
        tp = tc; tc = tn;
    }
    return o;
}

// ---------------------------------------------------------------------------
// Kernel D: fused elementwise pass (bandwidth carrier, b128 loads/stores).
// Per (b,l) row: xn = x + delta[b] (delta staged to LDS via TDM DMA);
// L2-normalize over D; Chebyshev; out = xn + 0.1*cheb(xn_norm).
// ---------------------------------------------------------------------------
__global__ void fused_elem(const float* __restrict__ x,
                           const float* __restrict__ delta,
                           const float* __restrict__ coeffs,
                           float* __restrict__ out) {
    const int row = blockIdx.x;              // 0..8191 == b*L + l
    const int b   = row >> 11;
    const int t   = threadIdx.x;             // 0..255

    __shared__ float sdelta[D_DIM];          // 8 KB staged delta row
    __shared__ float red[256];

#if USE_TDM
    if (t < 32) {                            // wave 0 issues one TDM DMA
        const unsigned long long ga = (unsigned long long)(delta + (size_t)b * D_DIM);
        const unsigned lds_addr = (unsigned)(unsigned long long)(void*)sdelta;
        u32x4 g0;                            // D# group 0 (ISA 8.3)
        g0[0] = 1u;                          // count=1, is_restore=0, gather off
        g0[1] = lds_addr;                    // lds_addr
        g0[2] = (unsigned)(ga & 0xFFFFFFFFu);          // global_addr[31:0]
        g0[3] = (unsigned)((ga >> 32) & 0x1FFFFFFu)    // global_addr[56:32]
              | (2u << 30);                  // type=2 ("image")
        i32x8 g1 = {};                       // D# group 1 (ISA 8.4)
        g1[0] = (int)(2u << 16);             // wg_mask=0, data_size=2 (4 bytes)
        g1[1] = (int)((D_DIM & 0xFFFF) << 16);         // tensor_dim0[15:0]
        g1[2] = (int)((D_DIM >> 16) | (1u << 16));     // dim0 hi | tensor_dim1=1
        g1[3] = (int)((unsigned)D_DIM << 16);          // dim1 hi=0 | tile_dim0=2048
        g1[4] = 0;                                     // tile_dim1=0, tile_dim2=0
        g1[5] = D_DIM;                                 // tensor_dim0_stride lo32
        g1[6] = 0; g1[7] = 0;
        i32x4 g2 = {}, g3 = {};              // groups 2/3 unused (<=2D tensor)
#if __clang_major__ >= 23
        i32x8 g4 = {};
        __builtin_amdgcn_tensor_load_to_lds(g0, g1, g2, g3, g4, 0);
#else
        __builtin_amdgcn_tensor_load_to_lds(g0, g1, g2, g3, 0);
#endif
        __builtin_amdgcn_s_wait_tensorcnt(0);
    }
    __syncthreads();
#else
    for (int j = t; j < D_DIM; j += 256) sdelta[j] = delta[(size_t)b * D_DIM + j];
    __syncthreads();
#endif

    const float4* xr4 = (const float4*)(x + (size_t)row * D_DIM);
    float4* or4 = (float4*)(out + (size_t)row * D_DIM);
    const float4* sd4 = (const float4*)sdelta;

    float4 v[2];
    float sq = 0.0f;
    #pragma unroll
    for (int j = 0; j < 2; ++j) {
        const int q = t + j * 256;           // float4 index 0..511
        float4 xv = xr4[q];                  // global_load_b128
        const float4 dv = sd4[q];            // ds_load_b128
        xv.x += dv.x; xv.y += dv.y; xv.z += dv.z; xv.w += dv.w;
        v[j] = xv;
        sq += xv.x * xv.x + xv.y * xv.y + xv.z * xv.z + xv.w * xv.w;
    }

    red[t] = sq;
    __syncthreads();
    #pragma unroll
    for (int s = 128; s >= 1; s >>= 1) {
        if (t < s) red[t] += red[t + s];
        __syncthreads();
    }
    const float rinv = rsqrtf(fmaxf(red[0], 1e-8f));

    float c[9];
    #pragma unroll
    for (int i = 0; i < 9; ++i) c[i] = coeffs[i];

    #pragma unroll
    for (int j = 0; j < 2; ++j) {
        float4 xv = v[j];
        float4 o;
        o.x = xv.x + 0.1f * cheb8(xv.x * rinv, c);
        o.y = xv.y + 0.1f * cheb8(xv.y * rinv, c);
        o.z = xv.z + 0.1f * cheb8(xv.z * rinv, c);
        o.w = xv.w + 0.1f * cheb8(xv.w * rinv, c);
        or4[t + j * 256] = o;                // global_store_b128
    }
}

// ---------------------------------------------------------------------------
extern "C" void kernel_launch(void* const* d_in, const int* in_sizes, int n_in,
                              void* d_out, int out_size, void* d_ws, size_t ws_size,
                              hipStream_t stream) {
    const float* x       = (const float*)d_in[0];   // [4,2048,2048]
    const float* j_upper = (const float*)d_in[1];   // [2048,2048]
    const float* r_diag  = (const float*)d_in[2];   // [2048]
    const float* h_proj  = (const float*)d_in[3];   // [2048,2048]
    const float* coeffs  = (const float*)d_in[4];   // [9]
    float* out = (float*)d_out;

    float* ws     = (float*)d_ws;
    float* part   = ws;                                     // 16*4*2048 floats
    float* x_mean = part + (size_t)S_SPLIT * B_DIM * D_DIM; // 8192
    float* g      = x_mean + (size_t)B_DIM * D_DIM;         // 8192
    float* delta  = g + (size_t)B_DIM * D_DIM;              // 8192

    colsum_partial<<<dim3(B_DIM, D_DIM / 1024, S_SPLIT), 256, 0, stream>>>(x, part);
    colsum_finish<<<(B_DIM * D_DIM / 4) / 256, 256, 0, stream>>>(part, x_mean);
    gemm1_wmma<<<D_DIM / 16, 32, 0, stream>>>(h_proj, x_mean, g);
    gemm2_wmma<<<D_DIM / 16, 32, 0, stream>>>(j_upper, r_diag, g, delta);
    fused_elem<<<B_DIM * L_DIM, 256, 0, stream>>>(x, delta, coeffs, out);
}
